// FoulModel_58308476010529
// MI455X (gfx1250) — matrix operand order repaired and checked by reading, not compiled, not run on hardware
//
#include <hip/hip_runtime.h>
#include <hip/hip_bf16.h>

// Problem constants (from reference): B=8, C=64, T=16, H=W=56, Cr=16.
#define B_   8
#define C_   64
#define T_   16
#define H_   56
#define W_   56
#define CR_  16
#define HW_  (H_*W_)
#define NKT  18           // K tiles: kt = r*2 + cihalf  (r = kh*3+kw in [0,9))
#define WC   58           // LDS cols: col 0 = w=-1 (zero), col 57 = w=56 (zero)
#define CP   66           // ci stride (64 + 2 pad -> 33-bank stride, conflict-free)
#define NTHR 224          // 7 waves: wave w owns N-tile w (16 pixels)

typedef __attribute__((ext_vector_type(16))) __bf16 v16bf;
typedef __attribute__((ext_vector_type(8)))  float  v8f;

__device__ __forceinline__ unsigned short f2bf(float f) {
  // round-to-nearest-even fp32 -> bf16
  unsigned u = __float_as_uint(f);
  u += 0x7FFFu + ((u >> 16) & 1u);
  return (unsigned short)(u >> 16);
}

__device__ __forceinline__ void atomicMaxF(float* addr, float v) {
  // float max via int/uint ordering trick (valid with -inf init, mixed signs)
  if (v >= 0.0f) atomicMax((int*)addr, __float_as_int(v));
  else           atomicMin((unsigned int*)addr, __float_as_uint(v));
}

// ---------------------------------------------------------------------------
// K mapping (shared by A-prep and B-gather): tile kt = r*2 + cihalf, where
// r = kh*3+kw. Within a tile, hardware K index Kidx (0..31) maps to input
// channel ci = cihalf*32 + Kidx, filter tap fixed at (kh,kw). This is a
// bijection onto k = ci*9 + r, and makes every B gather offset
//   [compile-time const(kt,s)] + [per-lane base + 16*hi]  (ci-innermost LDS)
// so the unrolled loop needs no per-slot address math at all.
// ---------------------------------------------------------------------------

// 1) Prep: Wconv (fp32 [64][64][3][3]) -> bf16 A-fragments, 16x32 bf16
//    A-layout (contiguous K order chosen here): lane L (M=L%16, hi=L/16), slot s:
//    Kidx = s + 16*hi. Flat [mtile][kt][lane][slot].
//    Also init pooled[B*C] = -inf.
__global__ void prep_kernel(const float* __restrict__ Wconv,
                            unsigned short* __restrict__ Aw,
                            float* __restrict__ pooled) {
  int idx = blockIdx.x * blockDim.x + threadIdx.x;
  if (idx < B_ * C_) pooled[idx] = -INFINITY;
  if (idx < 4 * NKT * 32 * 16) {
    int slot = idx & 15;
    int lane = (idx >> 4) & 31;
    int kt   = (idx >> 9) % NKT;
    int mt   = idx / (NKT * 512);
    int m    = mt * 16 + (lane & 15);
    int hi   = lane >> 4;
    int kidx = slot + 16 * hi;                        // A-layout K within tile
    int r    = kt >> 1;                               // filter tap
    int kh   = r / 3;
    int kw   = r - kh * 3;
    int ci   = (kt & 1) * 32 + kidx;                  // channel
    Aw[idx] = f2bf(Wconv[((m * C_ + ci) * 3 + kh) * 3 + kw]);
  }
}

// 2) Pooling pass: xt[b,c,t] = mean_{h,w}(x) + emb[view][c];
//    g[b,c] = mean_t xt[b,c,t]
__global__ void pool_kernel(const float* __restrict__ x,
                            const int* __restrict__ viewp,
                            const float* __restrict__ emb,
                            float* __restrict__ xt,
                            float* __restrict__ g) {
  __shared__ float red[256];
  int bc  = blockIdx.x;           // b*64 + c
  int tid = threadIdx.x;
  int view = *viewp;
  float e = emb[view * C_ + (bc & 63)];
  const float* xp = x + (size_t)bc * T_ * HW_;
  float gsum = 0.0f;
  for (int t = 0; t < T_; ++t) {
    float s = 0.0f;
    for (int i = tid; i < HW_; i += 256) s += xp[t * HW_ + i];
    red[tid] = s; __syncthreads();
    for (int st = 128; st > 0; st >>= 1) {
      if (tid < st) red[tid] += red[tid + st];
      __syncthreads();
    }
    float m = red[0] * (1.0f / HW_) + e;
    if (tid == 0) xt[bc * T_ + t] = m;
    gsum += m;
    __syncthreads();
  }
  if (tid == 0) g[bc] = gsum * (1.0f / T_);
}

// 3) RouteFuncMLP (tiny): alpha[b,c,t]. Single block, 512 threads.
__global__ void alpha_kernel(const float* __restrict__ xt,
                             const float* __restrict__ g,
                             const float* __restrict__ Wg, const float* __restrict__ bg,
                             const float* __restrict__ Wa, const float* __restrict__ ba,
                             const float* __restrict__ bn_g, const float* __restrict__ bn_b,
                             const float* __restrict__ bn_m, const float* __restrict__ bn_v,
                             const float* __restrict__ Wb,
                             float* __restrict__ alpha) {
  __shared__ float s_u[B_ * C_ * T_];
  __shared__ float s_h[B_ * CR_ * T_];
  __shared__ float s_gp[B_ * C_];
  int tid = threadIdx.x;                 // 512 threads
  {
    int b = tid >> 6, o = tid & 63;
    float acc = bg[o];
    for (int c = 0; c < C_; ++c) acc += Wg[o * C_ + c] * g[b * C_ + c];
    s_gp[tid] = acc;
  }
  __syncthreads();
  for (int idx = tid; idx < B_ * C_ * T_; idx += 512)
    s_u[idx] = xt[idx] + s_gp[idx >> 4];
  __syncthreads();
  for (int idx = tid; idx < B_ * CR_ * T_; idx += 512) {
    int b = idx >> 8, r = (idx >> 4) & 15, t = idx & 15;
    float acc = ba[r];
    for (int c = 0; c < C_; ++c) {
      const float* u = &s_u[(b * C_ + c) * T_];
      const float* w = &Wa[(r * C_ + c) * 3];
      if (t > 0)      acc += w[0] * u[t - 1];
                      acc += w[1] * u[t];
      if (t < T_ - 1) acc += w[2] * u[t + 1];
    }
    acc = (acc - bn_m[r]) * rsqrtf(bn_v[r] + 1e-5f) * bn_g[r] + bn_b[r];
    s_h[idx] = fmaxf(acc, 0.0f);
  }
  __syncthreads();
  for (int idx = tid; idx < B_ * C_ * T_; idx += 512) {
    int b = idx >> 10, c = (idx >> 4) & 63, t = idx & 15;
    float acc = 1.0f;
    for (int r = 0; r < CR_; ++r) {
      const float* hh = &s_h[(b * CR_ + r) * T_];
      const float* w  = &Wb[(c * CR_ + r) * 3];
      if (t > 0)      acc += w[0] * hh[t - 1];
                      acc += w[1] * hh[t];
      if (t < T_ - 1) acc += w[2] * hh[t + 1];
    }
    alpha[idx] = acc;
  }
}

// 4) Fused TAdaConv2d + max-pool. One block = (b, t, row-pair rp): 112 output
//    pixels = rows 2rp..2rp+1. 7 waves; wave w owns N-tile w and ALL 4
//    M-tiles -> each B fragment gathered exactly once. Implicit GEMM M=64,
//    N=112, K=576, v_wmma_f32_16x16x32_bf16, kt fully unrolled.
//    LDS tile layout [row 0..3][col 0..57][ci 0..65(pad)], pre-zero-padded,
//    so each B slot pair (2 adjacent ci) is one ds_load_b32 with an
//    immediate offset off a single per-lane base register.
//    B-fragment K order (contiguous, same as A-prep):
//    lane L (N=L%16, hi=L/16), slot s: Kidx = s + 16*hi.
__global__ __launch_bounds__(NTHR)
void tada_conv_kernel(const float* __restrict__ x,
                      const int* __restrict__ viewp,
                      const float* __restrict__ emb,
                      const float* __restrict__ alpha,
                      const unsigned short* __restrict__ Aw,
                      float* __restrict__ pooled) {
  __shared__ __align__(16) unsigned short rt[4 * WC * CP];   // ~30 KB
  __shared__ float s_scale[C_], s_bias[C_];
  __shared__ float s_pmax[C_];
  int rp  = blockIdx.x;      // 0..27
  int t   = blockIdx.y;      // 0..15
  int b   = blockIdx.z;      // 0..7
  int tid = threadIdx.x;

  if (tid < C_) {
    int view = *viewp;
    float al = alpha[(b * C_ + tid) * T_ + t];
    s_scale[tid] = al;                       // xs = (x + emb)*alpha
    s_bias[tid]  = emb[view * C_ + tid] * al;
    s_pmax[tid]  = -INFINITY;
  }
  __syncthreads();

  int h0 = rp * 2;
  // Stage rows h0-1 .. h0+2, 64 ch, scaled bf16. Coalesced along w; store
  // ci-innermost. Zeros in pad cols 0/57 and OOB halo rows.
  for (int i = tid; i < C_ * 4 * WC; i += NTHR) {
    int ci  = i / (4 * WC);
    int rem = i - ci * 4 * WC;
    int rr  = rem / WC;
    int wc  = rem - rr * WC;   // stored col; represents w = wc-1
    int hh  = h0 - 1 + rr;
    float v = 0.0f;
    if (hh >= 0 && hh < H_ && wc >= 1 && wc <= W_)
      v = x[((size_t)(b * C_ + ci) * T_ + t) * HW_ + hh * W_ + (wc - 1)]
              * s_scale[ci] + s_bias[ci];
    rt[(rr * WC + wc) * CP + ci] = f2bf(v);
  }
  __syncthreads();

  const unsigned* lds32 = (const unsigned*)rt;
  int wave = tid >> 5;          // = N-tile index (0..6)
  int lane = tid & 31;
  int hi   = lane >> 4;
  int ncol = lane & 15;
  int n    = wave * 16 + ncol;
  int nh   = (n >= W_) ? 1 : 0;           // which of the two output rows
  int w    = n - nh * W_;
  // per-lane dword base: ((nh*WC + w)*CP + 16*hi) / 2
  int base2h = (nh * WC + w) * 33 + 8 * hi;

  v8f acc[4] = {};
  const __bf16* Ab = (const __bf16*)Aw + lane * 16;

  #pragma unroll
  for (int kt = 0; kt < NKT; ++kt) {
    const int r   = kt >> 1;
    const int kh  = r / 3;                 // compile-time after unroll
    const int kw  = r - kh * 3;
    const int ch2 = (kt & 1) * 16;         // (cihalf*32)/2
    const int ib  = (kh * WC + kw) * 33 + ch2;   // compile-time dword offset
    union { v16bf v; unsigned u[8]; } bb;
    #pragma unroll
    for (int p = 0; p < 8; ++p)
      bb.u[p] = lds32[ib + p + base2h];    // ds_load_b32, imm offset
    #pragma unroll
    for (int mt = 0; mt < 4; ++mt) {
      v16bf a = *(const v16bf*)(Ab + (size_t)(mt * NKT + kt) * 512);
      acc[mt] = __builtin_amdgcn_wmma_f32_16x16x32_bf16(
          false, a, false, bb.v, (short)0, acc[mt], false, false);
    }
  }

  // Max over this wave's 16 columns. C/D layout: VGPR j, lane L ->
  // M = j + 8*(L/16), N = L%16. Reduce 16-lane groups, LDS max, then one
  // global atomic per (b, co).
  #pragma unroll
  for (int mt = 0; mt < 4; ++mt) {
    #pragma unroll
    for (int j = 0; j < 8; ++j) {
      float m = acc[mt][j];
      #pragma unroll
      for (int off = 8; off >= 1; off >>= 1)
        m = fmaxf(m, __shfl_xor(m, off, 32));
      if ((lane & 15) == 0)
        atomicMaxF(&s_pmax[mt * 16 + hi * 8 + j], m);
    }
  }
  __syncthreads();
  if (tid < C_) atomicMaxF(&pooled[b * C_ + tid], s_pmax[tid]);
}

// 5) Classifier: out[b,j] = pooled[b,:] . Wcls[j,:] + bcls[j]
__global__ void cls_kernel(const float* __restrict__ pooled,
                           const float* __restrict__ Wcls,
                           const float* __restrict__ bcls,
                           float* __restrict__ out) {
  int tid = threadIdx.x;
  if (tid < 80) {
    int b = tid / 10, j = tid - b * 10;
    float acc = bcls[j];
    for (int c = 0; c < C_; ++c) acc += pooled[b * C_ + c] * Wcls[j * C_ + c];
    out[tid] = acc;
  }
}

extern "C" void kernel_launch(void* const* d_in, const int* in_sizes, int n_in,
                              void* d_out, int out_size, void* d_ws, size_t ws_size,
                              hipStream_t stream) {
  const float* x     = (const float*)d_in[0];
  const int*   view  = (const int*)d_in[1];
  const float* emb   = (const float*)d_in[2];
  const float* Wg    = (const float*)d_in[3];
  const float* bg    = (const float*)d_in[4];
  const float* Wa    = (const float*)d_in[5];
  const float* ba    = (const float*)d_in[6];
  const float* bng   = (const float*)d_in[7];
  const float* bnb   = (const float*)d_in[8];
  const float* bnm   = (const float*)d_in[9];
  const float* bnv   = (const float*)d_in[10];
  const float* Wb    = (const float*)d_in[11];
  const float* Wconv = (const float*)d_in[12];
  const float* Wcls  = (const float*)d_in[13];
  const float* bcls  = (const float*)d_in[14];
  float* out = (float*)d_out;

  // Workspace layout (all re-initialized every call; ~143 KB total).
  float* alpha  = (float*)d_ws;            // 8192 f  [B][C][T]
  float* xt     = alpha + 8192;            // 8192 f  [B][C][T]
  float* g      = xt + 8192;               // 512 f   [B][C]
  float* pooled = g + 512;                 // 512 f   [B][C]
  unsigned short* Aw = (unsigned short*)(pooled + 512);  // 36864 bf16 fragments

  prep_kernel<<<144, 256, 0, stream>>>(Wconv, Aw, pooled);
  pool_kernel<<<B_ * C_, 256, 0, stream>>>(x, view, emb, xt, g);
  alpha_kernel<<<1, 512, 0, stream>>>(xt, g, Wg, bg, Wa, ba, bng, bnb, bnm, bnv,
                                      Wb, alpha);
  dim3 grid(H_ / 2, T_, B_);               // (28, 16, 8)
  tada_conv_kernel<<<grid, NTHR, 0, stream>>>(x, view, emb, alpha, Aw, pooled);
  cls_kernel<<<1, 128, 0, stream>>>(pooled, Wcls, bcls, out);
}
